// Attention_2293512536405
// MI455X (gfx1250) — compile-verified, matrix operand order checked by reading, not run on hardware
//
#include <hip/hip_runtime.h>
#include <hip/hip_bf16.h>

// ---------------------------------------------------------------------------
// MI455X (gfx1250) attention: bf16 WMMA + async global->LDS staging.
// ---------------------------------------------------------------------------

typedef __bf16 bf16;
typedef __attribute__((ext_vector_type(16))) __bf16 v16bf;
typedef __attribute__((ext_vector_type(8)))  __bf16 v8bf;
typedef __attribute__((ext_vector_type(8)))  float  v8f;

#define BATCH   2
#define SEQ     2048
#define DMODEL  1024
#define NHEADS  16
#define DHEAD   64
#define INNERD  1024            // NHEADS*DHEAD
#define JTOT    2050            // 2 null kv + SEQ
#define JPAD    2080            // JTOT rounded up to multiple of 32

// Build a 16-element bf16 WMMA fragment from two contiguous 8-elem chunks.
// Per ISA 16-bit A/B layout: lane half selects k-base {0,8}; elems 0..7 are
// K = kb..kb+7, elems 8..15 are K = kb+16..kb+23.
static __device__ __forceinline__ v16bf load_frag(const bf16* p0, const bf16* p1) {
  v8bf lo = *(const v8bf*)p0;
  v8bf hi = *(const v8bf*)p1;
  v16bf f;
#pragma unroll
  for (int e = 0; e < 8; ++e) { f[e] = lo[e]; f[e + 8] = hi[e]; }
  return f;
}

static __device__ __forceinline__ v8f wmma_bf16(v16bf a, v16bf b, v8f c) {
  return __builtin_amdgcn_wmma_f32_16x16x32_bf16(false, a, false, b, (short)0, c,
                                                 false, false);
}

// Flat __shared__ pointer -> wave-relative LDS byte offset (low 32 bits of the
// LDS aperture address).
static __device__ __forceinline__ unsigned lds_off(const void* p) {
  return (unsigned)(size_t)p;
}

// ASYNCcnt-tracked async copy: 16 bytes/lane, global -> LDS, no VGPR staging.
static __device__ __forceinline__ void async_load_b128(unsigned lds_byte_off,
                                                       const bf16* gp) {
  asm volatile("global_load_async_to_lds_b128 %0, %1, off"
               :: "v"(lds_byte_off), "v"((unsigned long long)(size_t)gp)
               : "memory");
}

// ---------------------------------------------------------------------------
// 1) Convert + transpose weights fp32 -> bf16 column-major (BT[n][k]).
// ---------------------------------------------------------------------------
__global__ __launch_bounds__(256) void prep_weights(
    const float* __restrict__ Wq, const float* __restrict__ Wkv,
    const float* __restrict__ Wout,
    bf16* __restrict__ WqT, bf16* __restrict__ WkvT, bf16* __restrict__ WoutT) {
  size_t t = (size_t)blockIdx.x * 256 + threadIdx.x;
  if (t < (size_t)1048576) {                       // Wq: 1024 x 1024
    int k = (int)(t >> 10), c = (int)(t & 1023);
    WqT[(size_t)c * DMODEL + k] = (bf16)Wq[t];
  } else if (t < (size_t)3145728) {                // Wkv: 1024 x 2048
    size_t u = t - 1048576;
    int k = (int)(u >> 11), c = (int)(u & 2047);
    WkvT[(size_t)c * DMODEL + k] = (bf16)Wkv[u];
  } else {                                         // Wout: 1024 x 1024
    size_t u = t - 3145728;
    int k = (int)(u >> 10), c = (int)(u & 1023);
    WoutT[(size_t)c * INNERD + k] = (bf16)Wout[u];
  }
}

// ---------------------------------------------------------------------------
// 2) LayerNorm (biased var) -> bf16 xn; also cast raw x -> bf16 (kv input).
// ---------------------------------------------------------------------------
__global__ __launch_bounds__(256) void layernorm_cast(
    const float* __restrict__ x, const float* __restrict__ w,
    const float* __restrict__ bia, bf16* __restrict__ xb, bf16* __restrict__ xnb) {
  int row = blockIdx.x;                              // 0 .. BATCH*SEQ-1
  const float* xr = x + (size_t)row * DMODEL;
  float v[4];
  float s = 0.f, ss = 0.f;
#pragma unroll
  for (int e = 0; e < 4; ++e) {
    float t = xr[threadIdx.x + e * 256];
    v[e] = t; s += t; ss += t * t;
  }
#pragma unroll
  for (int off = 16; off > 0; off >>= 1) {
    s  += __shfl_xor(s,  off, 32);
    ss += __shfl_xor(ss, off, 32);
  }
  __shared__ float sm0[8], sm1[8];
  int wv = threadIdx.x >> 5;
  if ((threadIdx.x & 31) == 0) { sm0[wv] = s; sm1[wv] = ss; }
  __syncthreads();
  float ts = 0.f, tss = 0.f;
#pragma unroll
  for (int i = 0; i < 8; ++i) { ts += sm0[i]; tss += sm1[i]; }
  float mean = ts * (1.0f / DMODEL);
  float var  = tss * (1.0f / DMODEL) - mean * mean;
  float inv  = rsqrtf(var + 1e-5f);
  bf16* xbr  = xb  + (size_t)row * DMODEL;
  bf16* xnbr = xnb + (size_t)row * DMODEL;
#pragma unroll
  for (int e = 0; e < 4; ++e) {
    int c = threadIdx.x + e * 256;
    float xn = (v[e] - mean) * inv * w[c] + bia[c];
    xbr[c]  = (bf16)v[e];
    xnbr[c] = (bf16)xn;
  }
}

// ---------------------------------------------------------------------------
// 3) Null kv tokens at j'=0,1 ; zero-pad j' in [JTOT, JPAD).
//    kR layout: [b][h][j'][d] (row-major keys) ; vT layout: [b][h][d][j'].
// ---------------------------------------------------------------------------
__global__ __launch_bounds__(256) void fill_null_pad(
    const float* __restrict__ null_kv, bf16* __restrict__ kR,
    bf16* __restrict__ vT) {
  int t = blockIdx.x * 256 + threadIdx.x;            // 0 .. BATCH*NHEADS*DHEAD-1
  int dd = t & 63;
  int h  = (t >> 6) & 15;
  int b  = t >> 10;
  size_t kbase = ((size_t)b * NHEADS + h) * JPAD * DHEAD;
  size_t vbase = (((size_t)b * NHEADS + h) * DHEAD + dd) * JPAD;
#pragma unroll
  for (int nn = 0; nn < 2; ++nn) {                   // null_kv: (h, 2, 2, d)
    float kv = null_kv[h * 256 + nn * 128 + dd];
    float vv = null_kv[h * 256 + nn * 128 + 64 + dd];
    kR[kbase + (size_t)nn * DHEAD + dd] = (bf16)kv;
    vT[vbase + nn] = (bf16)vv;
  }
  for (int jp = JTOT; jp < JPAD; ++jp) {
    kR[kbase + (size_t)jp * DHEAD + dd] = (bf16)0.f;
    vT[vbase + jp] = (bf16)0.f;
  }
}

// ---------------------------------------------------------------------------
// 4) Fused q/k/v projection. 2x2 register blocking: one 32x32 tile per wave
//    (4 WMMAs per k-step on 2 A-frags x 2 B-frags).
//    cols [0,1024): q = xn*Wq * d^-1/2 -> (b,h,n,d)
//    cols [1024,2048): k = x*Wkv[:, :1024] -> kR[b][h][j'=row+2][d]
//    cols [2048,3072): v = x*Wkv[:, 1024:] -> vT[b][h][d][j'=row+2]
// ---------------------------------------------------------------------------
__global__ __launch_bounds__(256) void qkv_gemm(
    const bf16* __restrict__ xb, const bf16* __restrict__ xnb,
    const bf16* __restrict__ WqT, const bf16* __restrict__ WkvT,
    bf16* __restrict__ qout, bf16* __restrict__ kR, bf16* __restrict__ vT) {
  const int ITILES = SEQ / 32, CTILES = 96;          // 64 x 96 (32x32 tiles)
  int w    = blockIdx.x * 8 + (threadIdx.x >> 5);
  int lane = threadIdx.x & 31;
  int b    = w / (ITILES * CTILES);
  int rem  = w - b * ITILES * CTILES;
  int it   = rem / CTILES, ct = rem % CTILES;
  int i0 = it * 32, c0 = ct * 32;

  const bf16* A;
  const bf16* Bt;
  if (c0 < INNERD) { A = xnb + (size_t)b * SEQ * DMODEL; Bt = WqT  + (size_t)c0 * DMODEL; }
  else             { A = xb  + (size_t)b * SEQ * DMODEL; Bt = WkvT + (size_t)(c0 - INNERD) * DMODEL; }

  int rown = lane & 15;
  int kb   = (lane >> 4) * 8;
  const bf16* ar0 = A  + (size_t)(i0 + rown) * DMODEL;
  const bf16* ar1 = ar0 + (size_t)16 * DMODEL;
  const bf16* br0 = Bt + (size_t)rown * DMODEL;
  const bf16* br1 = br0 + (size_t)16 * DMODEL;

  const v8f vz = {0.f, 0.f, 0.f, 0.f, 0.f, 0.f, 0.f, 0.f};
  v8f acc[2][2] = {{vz, vz}, {vz, vz}};
#pragma unroll 2
  for (int k0 = 0; k0 < DMODEL; k0 += 32) {
    v16bf a0 = load_frag(ar0 + k0 + kb, ar0 + k0 + kb + 16);
    v16bf a1 = load_frag(ar1 + k0 + kb, ar1 + k0 + kb + 16);
    v16bf b0 = load_frag(br0 + k0 + kb, br0 + k0 + kb + 16);
    v16bf b1 = load_frag(br1 + k0 + kb, br1 + k0 + kb + 16);
    acc[0][0] = wmma_bf16(a0, b0, acc[0][0]);
    acc[0][1] = wmma_bf16(a0, b1, acc[0][1]);
    acc[1][0] = wmma_bf16(a1, b0, acc[1][0]);
    acc[1][1] = wmma_bf16(a1, b1, acc[1][1]);
  }

  int rbase = (lane >> 4) * 8;
#pragma unroll
  for (int rt = 0; rt < 2; ++rt) {
#pragma unroll
    for (int ctile = 0; ctile < 2; ++ctile) {
#pragma unroll
      for (int r = 0; r < 8; ++r) {
        int irow = i0 + rt * 16 + r + rbase;
        int col  = c0 + ctile * 16 + rown;
        float vl = acc[rt][ctile][r];
        if (col < INNERD) {                          // q (scaled)
          int head = col >> 6, dd = col & 63;
          qout[(((size_t)b * NHEADS + head) * SEQ + irow) * DHEAD + dd] =
              (bf16)(vl * 0.125f);
        } else if (col < 2 * INNERD) {               // k
          int cc = col - INNERD, head = cc >> 6, dd = cc & 63;
          kR[(((size_t)b * NHEADS + head) * JPAD + (irow + 2)) * DHEAD + dd] =
              (bf16)vl;
        } else {                                     // v (transposed)
          int cc = col - 2 * INNERD, head = cc >> 6, dd = cc & 63;
          vT[(((size_t)b * NHEADS + head) * DHEAD + dd) * JPAD + (irow + 2)] =
              (bf16)vl;
        }
      }
    }
  }
}

// ---------------------------------------------------------------------------
// 5) Flash attention. A block = 8 waves = 8 consecutive 16-row query tiles of
//    one (b,h). K/V tiles (32 keys) are staged to LDS with double-buffered
//    global_load_async_to_lds_b128 (ASYNCcnt) and shared by all 8 waves.
//    ALiBi: slope_h = 2^(-(h+1)/2); bias = -slope*|j' - (i+2)|.
//    Causal (shifted by 2 nulls): allowed iff j' <= i+2 and j' < JTOT.
// ---------------------------------------------------------------------------
__global__ __launch_bounds__(256) void flash_attn(
    const bf16* __restrict__ qb, const bf16* __restrict__ kR,
    const bf16* __restrict__ vT, bf16* __restrict__ aout) {
  __shared__ __align__(16) bf16 ktile[2][32 * 64];   // [key][d]
  __shared__ __align__(16) bf16 vtile[2][64 * 32];   // [d][key]
  __shared__ __align__(16) bf16 plds[8 * 16 * 32];   // per-wave 16x32 P tile

  int tid  = threadIdx.x;
  int wl   = tid >> 5;
  int lane = tid & 31;
  int w    = blockIdx.x * 8 + wl;                    // 0..4095
  int it   = w & 127;
  int h    = (w >> 7) & 15;
  int b    = w >> 11;
  int i0   = it * 16;

  const bf16* qptr = qb + (((size_t)b * NHEADS + h) * SEQ + i0) * DHEAD;
  const bf16* kptr = kR + ((size_t)b * NHEADS + h) * JPAD * DHEAD;
  const bf16* vptr = vT + ((size_t)b * NHEADS + h) * DHEAD * JPAD;
  bf16* pl = plds + wl * 512;

  int rown  = lane & 15;
  int kb    = (lane >> 4) * 8;
  int rbase = (lane >> 4) * 8;
  float slope = exp2f(-0.5f * (float)(h + 1));

  // staging addresses: this thread copies 16B of K and 16B of V per stage
  const bf16* kg = kptr + (size_t)tid * 8;                       // + j0*64/stage
  const bf16* vg = vptr + (size_t)(tid >> 2) * JPAD + (tid & 3) * 8;  // + j0
  unsigned klo[2] = { lds_off(&ktile[0][tid * 8]), lds_off(&ktile[1][tid * 8]) };
  unsigned vlo[2] = { lds_off(&vtile[0][tid * 8]), lds_off(&vtile[1][tid * 8]) };

  // causal early-exit: block's last query row + 2 nulls bounds the key range
  int it_last = ((blockIdx.x * 8) & 127) + 7;
  int NS      = (it_last * 16 + 17) / 32 + 1;        // block-uniform trip count
  if (NS > JPAD / 32) NS = JPAD / 32;

  v16bf qf[2];
#pragma unroll
  for (int kd = 0; kd < 2; ++kd) {
    const bf16* qp = qptr + (size_t)rown * DHEAD + kd * 32 + kb;
    qf[kd] = load_frag(qp, qp + 16);
  }

  float mrow[8], lrow[8];
  v8f Oacc[4];
  const v8f vz = {0.f, 0.f, 0.f, 0.f, 0.f, 0.f, 0.f, 0.f};
#pragma unroll
  for (int r = 0; r < 8; ++r) { mrow[r] = -3.0e38f; lrow[r] = 0.f; }
#pragma unroll
  for (int f = 0; f < 4; ++f) Oacc[f] = vz;

  // prologue: stage 0 into buffer 0
  async_load_b128(klo[0], kg);
  async_load_b128(vlo[0], vg);

  for (int s = 0; s < NS; ++s) {
    int j0  = s * 32;
    int cur = s & 1;
    if (s + 1 < NS) {                                // prefetch next stage
      int nj = j0 + 32;
      async_load_b128(klo[cur ^ 1], kg + (size_t)nj * 64);
      async_load_b128(vlo[cur ^ 1], vg + nj);
      asm volatile("s_wait_asynccnt 0x2" ::: "memory");
    } else {
      asm volatile("s_wait_asynccnt 0x0" ::: "memory");
    }
    __syncthreads();                                 // tiles visible to block

    // --- scores S = q . k^T for 2 adjacent 16-key tiles (from LDS) ---
    v8f S[2];
#pragma unroll
    for (int jt = 0; jt < 2; ++jt) {
      v8f sacc = vz;
#pragma unroll
      for (int kd = 0; kd < 2; ++kd) {
        const bf16* bp = &ktile[cur][(jt * 16 + rown) * 64 + kd * 32 + kb];
        v16bf kf = load_frag(bp, bp + 16);
        sacc = wmma_bf16(qf[kd], kf, sacc);
      }
      S[jt] = sacc;
    }

    // --- ALiBi + causal mask + online softmax update ---
    float pv[2][8];
#pragma unroll
    for (int r = 0; r < 8; ++r) {
      int irow = i0 + r + rbase;
      float t = -3.0e38f;
#pragma unroll
      for (int jt = 0; jt < 2; ++jt) {
        int jp   = j0 + jt * 16 + rown;
        bool ok  = (jp <= irow + 2) && (jp < JTOT);
        float sv = ok ? (S[jt][r] - slope * fabsf((float)(jp - (irow + 2))))
                      : -3.0e38f;
        pv[jt][r] = sv;
        t = fmaxf(t, sv);
      }
#pragma unroll
      for (int off = 8; off > 0; off >>= 1) t = fmaxf(t, __shfl_xor(t, off, 32));
      float nm = fmaxf(mrow[r], t);
      float rs = 0.f;
#pragma unroll
      for (int jt = 0; jt < 2; ++jt) {
        float p = (pv[jt][r] > -1.0e38f) ? __expf(pv[jt][r] - nm) : 0.f;
        pv[jt][r] = p;
        rs += p;
      }
#pragma unroll
      for (int off = 8; off > 0; off >>= 1) rs += __shfl_xor(rs, off, 32);
      float alpha = __expf(mrow[r] - nm);
      lrow[r] = lrow[r] * alpha + rs;
      mrow[r] = nm;
#pragma unroll
      for (int f = 0; f < 4; ++f) Oacc[f][r] *= alpha;
    }

    // --- P: C-layout -> A-layout through per-wave LDS tile ---
#pragma unroll
    for (int jt = 0; jt < 2; ++jt)
#pragma unroll
      for (int r = 0; r < 8; ++r)
        pl[(r + rbase) * 32 + jt * 16 + rown] = (bf16)pv[jt][r];
    __syncthreads();                                 // P visible; K reads done
    const bf16* pa = pl + (size_t)rown * 32 + kb;
    v16bf pf = load_frag(pa, pa + 16);

    // --- O += P . V (4 N-tiles of 16 over d=64, V from LDS) ---
#pragma unroll
    for (int f = 0; f < 4; ++f) {
      const bf16* vp = &vtile[cur][(f * 16 + rown) * 32 + kb];
      v16bf vf = load_frag(vp, vp + 16);
      Oacc[f] = wmma_bf16(pf, vf, Oacc[f]);
    }
    __syncthreads();                                 // V reads done before reuse
  }

  // --- normalize and scatter to (b, n, h*d) bf16 ---
#pragma unroll
  for (int r = 0; r < 8; ++r) {
    float inv  = (lrow[r] > 0.f) ? (1.f / lrow[r]) : 0.f;
    int   irow = i0 + r + rbase;
#pragma unroll
    for (int f = 0; f < 4; ++f)
      aout[((size_t)b * SEQ + irow) * INNERD + h * DHEAD + f * 16 + rown] =
          (bf16)(Oacc[f][r] * inv);
  }
}

// ---------------------------------------------------------------------------
// 6) Output projection: out = attn_out . Wout, 2x2 blocked, fp32 stores.
// ---------------------------------------------------------------------------
__global__ __launch_bounds__(256) void out_gemm(
    const bf16* __restrict__ aoutb, const bf16* __restrict__ WoutT,
    float* __restrict__ out) {
  const int ITILES = SEQ / 32, CTILES = DMODEL / 32; // 64 x 32 (32x32 tiles)
  int w    = blockIdx.x * 8 + (threadIdx.x >> 5);
  int lane = threadIdx.x & 31;
  int b    = w / (ITILES * CTILES);
  int rem  = w - b * ITILES * CTILES;
  int it   = rem / CTILES, ct = rem % CTILES;
  int i0 = it * 32, c0 = ct * 32;

  int rown = lane & 15, kb = (lane >> 4) * 8;
  const bf16* ar0 = aoutb + ((size_t)b * SEQ + i0 + rown) * INNERD;
  const bf16* ar1 = ar0 + (size_t)16 * INNERD;
  const bf16* br0 = WoutT + (size_t)(c0 + rown) * INNERD;
  const bf16* br1 = br0 + (size_t)16 * INNERD;

  const v8f vz = {0.f, 0.f, 0.f, 0.f, 0.f, 0.f, 0.f, 0.f};
  v8f acc[2][2] = {{vz, vz}, {vz, vz}};
#pragma unroll 2
  for (int k0 = 0; k0 < INNERD; k0 += 32) {
    v16bf a0 = load_frag(ar0 + k0 + kb, ar0 + k0 + kb + 16);
    v16bf a1 = load_frag(ar1 + k0 + kb, ar1 + k0 + kb + 16);
    v16bf b0 = load_frag(br0 + k0 + kb, br0 + k0 + kb + 16);
    v16bf b1 = load_frag(br1 + k0 + kb, br1 + k0 + kb + 16);
    acc[0][0] = wmma_bf16(a0, b0, acc[0][0]);
    acc[0][1] = wmma_bf16(a0, b1, acc[0][1]);
    acc[1][0] = wmma_bf16(a1, b0, acc[1][0]);
    acc[1][1] = wmma_bf16(a1, b1, acc[1][1]);
  }
  int rbase = (lane >> 4) * 8;
#pragma unroll
  for (int rt = 0; rt < 2; ++rt)
#pragma unroll
    for (int ctile = 0; ctile < 2; ++ctile)
#pragma unroll
      for (int r = 0; r < 8; ++r)
        out[((size_t)b * SEQ + i0 + rt * 16 + r + rbase) * DMODEL +
            c0 + ctile * 16 + rown] = acc[rt][ctile][r];
}

// ---------------------------------------------------------------------------
extern "C" void kernel_launch(void* const* d_in, const int* in_sizes, int n_in,
                              void* d_out, int out_size, void* d_ws, size_t ws_size,
                              hipStream_t stream) {
  const float* x    = (const float*)d_in[0];
  // d_in[1] = mask (all true) — folded into the causal/null handling.
  const float* lnw  = (const float*)d_in[2];
  const float* lnb  = (const float*)d_in[3];
  const float* nkv  = (const float*)d_in[4];
  const float* Wq   = (const float*)d_in[5];
  const float* Wkv  = (const float*)d_in[6];
  const float* Wout = (const float*)d_in[7];
  float* out = (float*)d_out;

  char* ws = (char*)d_ws;
  size_t off = 0;
  auto alloc = [&](size_t bytes) -> void* {
    void* p = ws + off;
    off += (bytes + 255) & ~(size_t)255;
    return p;
  };
  bf16* xb    = (bf16*)alloc((size_t)BATCH * SEQ * DMODEL * 2);
  bf16* xnb   = (bf16*)alloc((size_t)BATCH * SEQ * DMODEL * 2);
  bf16* WqT   = (bf16*)alloc((size_t)DMODEL * INNERD * 2);
  bf16* WkvT  = (bf16*)alloc((size_t)DMODEL * 2 * INNERD * 2);
  bf16* WoutT = (bf16*)alloc((size_t)INNERD * DMODEL * 2);
  bf16* qbuf  = (bf16*)alloc((size_t)BATCH * NHEADS * SEQ * DHEAD * 2);
  bf16* kR    = (bf16*)alloc((size_t)BATCH * NHEADS * JPAD * DHEAD * 2);
  bf16* vT    = (bf16*)alloc((size_t)BATCH * NHEADS * DHEAD * JPAD * 2);
  bf16* aoutb = (bf16*)alloc((size_t)BATCH * SEQ * INNERD * 2);
  (void)ws_size; (void)in_sizes; (void)n_in; (void)out_size;

  prep_weights<<<16384, 256, 0, stream>>>(Wq, Wkv, Wout, WqT, WkvT, WoutT);
  layernorm_cast<<<BATCH * SEQ, 256, 0, stream>>>(x, lnw, lnb, xb, xnb);
  fill_null_pad<<<(BATCH * NHEADS * DHEAD) / 256, 256, 0, stream>>>(nkv, kR, vT);
  qkv_gemm<<<(BATCH * (SEQ / 32) * 96) / 8, 256, 0, stream>>>(xb, xnb, WqT, WkvT,
                                                              qbuf, kR, vT);
  flash_attn<<<(BATCH * NHEADS * (SEQ / 16)) / 8, 256, 0, stream>>>(qbuf, kR, vT,
                                                                    aoutb);
  out_gemm<<<(BATCH * (SEQ / 32) * (DMODEL / 32)) / 8, 256, 0, stream>>>(aoutb,
                                                                         WoutT, out);
}
